// GroupQueryAtention_11742440587282
// MI455X (gfx1250) — compile-verified, matrix operand order checked by reading, not run on hardware
//
#include <hip/hip_runtime.h>
#include <hip/hip_bf16.h>

typedef __bf16 bf16_t;
typedef bf16_t bf16x16 __attribute__((ext_vector_type(16)));
typedef float  f32x8  __attribute__((ext_vector_type(8)));
typedef unsigned int u32x4 __attribute__((ext_vector_type(4)));
typedef int i32x8 __attribute__((ext_vector_type(8)));
typedef int i32x4 __attribute__((ext_vector_type(4)));

#define WMMA_BF16(a, b, c) \
  __builtin_amdgcn_wmma_f32_16x16x32_bf16(false, (a), false, (b), (short)0, (c), false, false)

union FragU { uint4 u[2]; bf16x16 v; };

// Load one 16x32 bf16 A/Bt fragment per the CDNA5 layout:
// lane row = l&15, koff = 8*(l>>4); elements [koff..koff+7] and [16+koff..16+koff+7]
#define LOAD_FRAG(dst, rowptr, koff_)                          \
  do {                                                         \
    FragU _f;                                                  \
    _f.u[0] = *(const uint4*)((rowptr) + (koff_));             \
    _f.u[1] = *(const uint4*)((rowptr) + 16 + (koff_));        \
    (dst) = _f.v;                                              \
  } while (0)

#define HAVE_TDM (__has_builtin(__builtin_amdgcn_tensor_load_to_lds) && \
                  __has_builtin(__builtin_amdgcn_s_wait_tensorcnt))

#if HAVE_TDM
// Issue one TDM 2D tile load: tile_w x tile_h elements (2 bytes each) from a
// row-major tensor with row stride `stride_elems`, into LDS at `lds_off`.
// D# bit layout per CDNA5 ISA §8.3/8.4 (group0: flags/lds/global/type,
// group1: data_size + dims + tile dims + dim0 stride). This toolchain exposes
// the 6-arg builtin: (g0 u32x4, g1 i32x8, g2 i32x4, g3 i32x4, i32x8, cpol).
static __device__ inline void tdm_load_tile_2d(const void* gaddr, unsigned lds_off,
                                               unsigned tile_w, unsigned tile_h,
                                               unsigned long long stride_elems) {
  unsigned long long ga = (unsigned long long)gaddr;
  u32x4 g0;
  g0[0] = 1u;                                              // count=1 (valid), user D#
  g0[1] = lds_off;                                         // lds_addr (bytes)
  g0[2] = (unsigned)(ga & 0xffffffffu);                    // global_addr[31:0]
  g0[3] = (unsigned)((ga >> 32) & 0x01ffffffu) | (2u << 30); // addr[56:32] | type=2
  i32x8 g1;
  g1[0] = (int)(1u << 16);                                 // data_size=1 -> 2 bytes
  g1[1] = (int)((tile_w & 0xffffu) << 16);                 // tensor_dim0[15:0] (=tile_w)
  g1[2] = (int)(((tile_w >> 16) & 0xffffu) | ((tile_h & 0xffffu) << 16)); // dim0 hi | dim1 lo
  g1[3] = (int)(((tile_h >> 16) & 0xffffu) | ((tile_w & 0xffffu) << 16)); // dim1 hi | tile_dim0
  g1[4] = (int)(tile_h & 0xffffu);                         // tile_dim1; tile_dim2=0
  g1[5] = (int)(unsigned)(stride_elems & 0xffffffffu);     // tensor_dim0_stride[31:0]
  g1[6] = (int)(unsigned)((stride_elems >> 32) & 0xffffu); // stride[47:32]; dim1_stride=0
  g1[7] = 0;
  i32x4 z4 = {0, 0, 0, 0};
  i32x8 z8 = {0, 0, 0, 0, 0, 0, 0, 0};
  __builtin_amdgcn_tensor_load_to_lds(g0, g1, z4, z4, z8, 0);
}
#endif

// ---------------------------------------------------------------------------
// fp32 -> bf16 conversion kernels
// ---------------------------------------------------------------------------
__global__ void cvt_bf16_kernel(const float* __restrict__ src, bf16_t* __restrict__ dst, int n) {
  int i = blockIdx.x * blockDim.x + threadIdx.x;
  if (i < n) dst[i] = (bf16_t)src[i];
}

// src is K x N row-major fp32; dst is N x K row-major bf16 (transposed)
__global__ void cvt_bf16_T_kernel(const float* __restrict__ src, bf16_t* __restrict__ dst,
                                  int K, int N) {
  int i = blockIdx.x * blockDim.x + threadIdx.x;
  if (i < K * N) {
    int k = i / N, n = i - k * N;
    dst[(size_t)n * K + k] = (bf16_t)src[i];
  }
}

// ---------------------------------------------------------------------------
// WMMA GEMM: C = A(MxK) * Bt^T (Bt is NxK row-major), bf16 in, f32 accumulate.
// Block tile 128x128, 8 waves of 32(M)x64(N), K-step 32.
// Tiles staged into LDS by the Tensor Data Mover (wave 0 issues two D#s per
// K-step, waits TENSORcnt, then the barrier releases the WMMA pipeline).
// mode 0: RoPE epilogue + head-major bf16 store (Q/K)
// mode 1: head-major bf16 store (V)
// mode 2: plain row-major f32 store (final projection)
// ---------------------------------------------------------------------------
__global__ __launch_bounds__(256)
void gemm_bf16_wmma(const bf16_t* __restrict__ A, const bf16_t* __restrict__ Bt,
                    int M, int N, int K, int mode,
                    const float* __restrict__ angles,
                    bf16_t* __restrict__ out_bf, float* __restrict__ out_f,
                    int heads, int Tdim, int HDdim) {
  __shared__ __align__(16) bf16_t As[128][32];
  __shared__ __align__(16) bf16_t Bs[128][32];

  int tid = threadIdx.x;
  int m0 = blockIdx.y * 128, n0 = blockIdx.x * 128;
  int w = tid >> 5, lane = tid & 31;
  int wm = (w >> 1) * 32, wn = (w & 1) * 64;
  int lr = lane & 15, hw = lane >> 4, koff = hw * 8;

  f32x8 acc[2][4] = {};

  for (int k0 = 0; k0 < K; k0 += 32) {
#if HAVE_TDM
    if (w == 0) {
      tdm_load_tile_2d(&A [(size_t)m0 * K + k0], (unsigned)(size_t)(void*)As,
                       32u, 128u, (unsigned long long)K);
      tdm_load_tile_2d(&Bt[(size_t)n0 * K + k0], (unsigned)(size_t)(void*)Bs,
                       32u, 128u, (unsigned long long)K);
      __builtin_amdgcn_s_wait_tensorcnt(0);
    }
#else
    // cooperative tile load: 2 x uint4 (16 bf16) per thread per matrix
    for (int r = 0; r < 2; ++r) {
      int idx = tid + (r << 8);           // 0..511
      int row = idx >> 2;                 // 0..127
      int kc  = (idx & 3) << 3;           // 0,8,16,24
      *(uint4*)&As[row][kc] = *(const uint4*)&A [(size_t)(m0 + row) * K + k0 + kc];
      *(uint4*)&Bs[row][kc] = *(const uint4*)&Bt[(size_t)(n0 + row) * K + k0 + kc];
    }
#endif
    if (k0 + 32 < K) {
      __builtin_prefetch(&A [(size_t)(m0 + (tid >> 1)) * K + k0 + 32], 0, 0);
      __builtin_prefetch(&Bt[(size_t)(n0 + (tid >> 1)) * K + k0 + 32], 0, 0);
    }
    __syncthreads();

    bf16x16 af[2], bfv[4];
    LOAD_FRAG(af[0], &As[wm + lr][0], koff);
    LOAD_FRAG(af[1], &As[wm + 16 + lr][0], koff);
    for (int j = 0; j < 4; ++j) LOAD_FRAG(bfv[j], &Bs[wn + 16 * j + lr][0], koff);

    for (int i = 0; i < 2; ++i)
      for (int j = 0; j < 4; ++j)
        acc[i][j] = WMMA_BF16(af[i], bfv[j], acc[i][j]);
    __syncthreads();
  }

  // Epilogue. C-layout: VGPR v, lane l -> row = v + 8*(l>>4), col = l&15.
  for (int i = 0; i < 2; ++i)
    for (int j = 0; j < 4; ++j)
      for (int v = 0; v < 8; ++v) {
        int row = m0 + wm + 16 * i + v + 8 * hw;
        int col = n0 + wn + 16 * j + lr;
        float val = acc[i][j][v];
        if (mode == 2) {
          out_f[(size_t)row * N + col] = val;
          continue;
        }
        int bb = row / Tdim, t = row - bb * Tdim;
        int hh = col / HDdim, d = col - hh * HDdim;
        if (mode == 0) {
          // RoPE: pair columns (2i, 2i+1) are adjacent lanes within a half-wave
          float other = __shfl_xor(val, 1, 32);
          float ang = angles[(size_t)t * (HDdim >> 1) + (d >> 1)];
          float cs = __cosf(ang), sn = __sinf(ang);
          val = ((d & 1) == 0) ? (cs * val - sn * other) : (sn * other + cs * val);
        }
        out_bf[((size_t)(bb * heads + hh) * Tdim + t) * HDdim + d] = (bf16_t)val;
      }
}

// ---------------------------------------------------------------------------
// Flash attention. Reference quirk: scores[i,j] = k_i . q_j, causal mask j<=i,
// softmax over j, out_i = sum_j w[i,j] v_j. So A-matrix = K rows, B = Q rows.
// Grid (T/64, NH, B), 128 threads = 4 waves, wave w owns rows i0+16w..+15.
// ---------------------------------------------------------------------------
#define AT_NH  16
#define AT_NKV 4
#define AT_HD  128

__global__ __launch_bounds__(128)
void attn_fwd(const bf16_t* __restrict__ qbf, const bf16_t* __restrict__ kbf,
              const bf16_t* __restrict__ vbf, bf16_t* __restrict__ obf, int Tdim) {
  __shared__ __align__(16) bf16_t VT[AT_HD][32];      // V^T tile: [d][j_local]
  __shared__ __align__(16) bf16_t Pb[4][16][32];      // per-wave P relayout buffer

  int tid = threadIdx.x, w = tid >> 5, lane = tid & 31;
  int lr = lane & 15, hw = lane >> 4, koff = hw * 8;
  int b = blockIdx.z, h = blockIdx.y, hk = h & (AT_NKV - 1);
  int i0 = blockIdx.x * 64;
  int iw = i0 + 16 * w;

  // Resident A-fragments: this wave's 16 K-rows, 4 chunks of 32 over d
  const bf16_t* krow = kbf + ((size_t)(b * AT_NKV + hk) * Tdim + (iw + lr)) * AT_HD;
  bf16x16 ak[4];
  for (int c = 0; c < 4; ++c) LOAD_FRAG(ak[c], krow + c * 32, koff);

  f32x8 o[8] = {};
  float mrow[8], lrow[8];
  for (int v = 0; v < 8; ++v) { mrow[v] = -3.0e38f; lrow[v] = 0.0f; }

  const bf16_t* qbase = qbf + (size_t)(b * AT_NH + h) * Tdim * AT_HD;
  const bf16_t* vbase = vbf + (size_t)(b * AT_NKV + hk) * Tdim * AT_HD;

  int ntiles = (i0 + 64) >> 5;   // causal: j tiles of 32 up through the diagonal
  for (int jt = 0; jt < ntiles; ++jt) {
    int j0 = jt << 5;
    __syncthreads();             // previous PV reads of VT complete

    // cooperative V^T tile: 32 j-rows x 128 d, transposed into LDS
    for (int ii = 0; ii < 8; ++ii) {
      int idx = tid + (ii << 7);             // 0..1023 quads
      int jr = idx >> 5;                     // 0..31
      int c4 = (idx & 31) << 2;              // 0..124
      const bf16_t* vp = vbase + (size_t)(j0 + jr) * AT_HD + c4;
      bf16_t v0 = vp[0], v1 = vp[1], v2 = vp[2], v3 = vp[3];
      VT[c4 + 0][jr] = v0; VT[c4 + 1][jr] = v1;
      VT[c4 + 2][jr] = v2; VT[c4 + 3][jr] = v3;
    }

    // scores: S(16x32) = K_rows x Q^T, two 16-col subtiles, K-dim = d = 128
    f32x8 s0 = {}, s1 = {};
    for (int c = 0; c < 4; ++c) {
      bf16x16 bq0, bq1;
      LOAD_FRAG(bq0, qbase + (size_t)(j0 + lr) * AT_HD + c * 32, koff);
      LOAD_FRAG(bq1, qbase + (size_t)(j0 + 16 + lr) * AT_HD + c * 32, koff);
      s0 = WMMA_BF16(ak[c], bq0, s0);
      s1 = WMMA_BF16(ak[c], bq1, s1);
    }

    // causal mask + online softmax; row r lives at VGPR v, half-wave hw,
    // 16 column values across lanes -> xor-reduce {1,2,4,8} stays in-half.
    for (int v = 0; v < 8; ++v) {
      int ig = iw + v + 8 * hw;
      float a0 = s0[v], a1 = s1[v];
      if (j0 + lr > ig)      a0 = -3.0e38f;
      if (j0 + 16 + lr > ig) a1 = -3.0e38f;
      float rmax = fmaxf(a0, a1);
      rmax = fmaxf(rmax, __shfl_xor(rmax, 1, 32));
      rmax = fmaxf(rmax, __shfl_xor(rmax, 2, 32));
      rmax = fmaxf(rmax, __shfl_xor(rmax, 4, 32));
      rmax = fmaxf(rmax, __shfl_xor(rmax, 8, 32));
      float mnew  = fmaxf(mrow[v], rmax);
      float scale = __expf(mrow[v] - mnew);
      float p0 = __expf(a0 - mnew), p1 = __expf(a1 - mnew);
      float rs = p0 + p1;
      rs += __shfl_xor(rs, 1, 32);
      rs += __shfl_xor(rs, 2, 32);
      rs += __shfl_xor(rs, 4, 32);
      rs += __shfl_xor(rs, 8, 32);
      lrow[v] = lrow[v] * scale + rs;
      mrow[v] = mnew;
      for (int c8 = 0; c8 < 8; ++c8) o[c8][v] *= scale;
      int ml = v + 8 * hw;
      Pb[w][ml][lr]      = (bf16_t)p0;
      Pb[w][ml][16 + lr] = (bf16_t)p1;
    }
    __syncthreads();             // VT + P visible

    // PV: out(16x128) += P(16x32) x V(32x128); one WMMA per 16-wide d-subtile
    bf16x16 pf;
    LOAD_FRAG(pf, &Pb[w][lr][0], koff);
    for (int c8 = 0; c8 < 8; ++c8) {
      bf16x16 bv;
      LOAD_FRAG(bv, &VT[16 * c8 + lr][0], koff);
      o[c8] = WMMA_BF16(pf, bv, o[c8]);
    }
  }

  // normalize and store (B, T, NH*HD) bf16 for the output projection
  for (int v = 0; v < 8; ++v) lrow[v] = 1.0f / lrow[v];
  for (int c8 = 0; c8 < 8; ++c8)
    for (int v = 0; v < 8; ++v) {
      int ig = iw + v + 8 * hw;
      int col = h * AT_HD + 16 * c8 + lr;
      obf[(size_t)(b * Tdim + ig) * (AT_NH * AT_HD) + col] = (bf16_t)(o[c8][v] * lrow[v]);
    }
}

// ---------------------------------------------------------------------------
// Host launcher
// ---------------------------------------------------------------------------
extern "C" void kernel_launch(void* const* d_in, const int* in_sizes, int n_in,
                              void* d_out, int out_size, void* d_ws, size_t ws_size,
                              hipStream_t stream) {
  const int B = 2, T = 1024, C = 2048, NH = 16, NKV = 4, HD = 128;
  const int KVC = NKV * HD;     // 512
  const int M = B * T;          // 2048

  const float* x      = (const float*)d_in[0];
  const float* angles = (const float*)d_in[1];
  const float* wq     = (const float*)d_in[2];
  const float* wk     = (const float*)d_in[3];
  const float* wv     = (const float*)d_in[4];
  const float* wo     = (const float*)d_in[5];
  float* outp = (float*)d_out;

  char* ws = (char*)d_ws;
  const size_t MB = 1u << 20;
  bf16_t* xbf = (bf16_t*)(ws + 0 * MB);    // 8 MB  (2048x2048)
  bf16_t* wqT = (bf16_t*)(ws + 8 * MB);    // 8 MB  (2048x2048 transposed)
  bf16_t* wkT = (bf16_t*)(ws + 16 * MB);   // 2 MB  (512x2048)
  bf16_t* wvT = (bf16_t*)(ws + 18 * MB);   // 2 MB
  bf16_t* woT = (bf16_t*)(ws + 20 * MB);   // 8 MB
  bf16_t* qbf = (bf16_t*)(ws + 28 * MB);   // 8 MB  (B,NH,T,HD)
  bf16_t* kbf = (bf16_t*)(ws + 36 * MB);   // 2 MB  (B,NKV,T,HD)
  bf16_t* vbf = (bf16_t*)(ws + 38 * MB);   // 2 MB
  bf16_t* abf = (bf16_t*)(ws + 40 * MB);   // 8 MB  (B*T, C)

  int n;
  n = M * C;
  cvt_bf16_kernel<<<(n + 255) / 256, 256, 0, stream>>>(x, xbf, n);
  n = C * C;
  cvt_bf16_T_kernel<<<(n + 255) / 256, 256, 0, stream>>>(wq, wqT, C, C);
  n = C * KVC;
  cvt_bf16_T_kernel<<<(n + 255) / 256, 256, 0, stream>>>(wk, wkT, C, KVC);
  cvt_bf16_T_kernel<<<(n + 255) / 256, 256, 0, stream>>>(wv, wvT, C, KVC);
  n = C * C;
  cvt_bf16_T_kernel<<<(n + 255) / 256, 256, 0, stream>>>(wo, woT, C, C);

  dim3 blk(256);
  // Q projection + RoPE -> (B,NH,T,HD)
  gemm_bf16_wmma<<<dim3(C / 128, M / 128), blk, 0, stream>>>(
      xbf, wqT, M, C, C, /*mode=*/0, angles, qbf, nullptr, NH, T, HD);
  // K projection + RoPE -> (B,NKV,T,HD)
  gemm_bf16_wmma<<<dim3(KVC / 128, M / 128), blk, 0, stream>>>(
      xbf, wkT, M, KVC, C, /*mode=*/0, angles, kbf, nullptr, NKV, T, HD);
  // V projection -> (B,NKV,T,HD)
  gemm_bf16_wmma<<<dim3(KVC / 128, M / 128), blk, 0, stream>>>(
      xbf, wvT, M, KVC, C, /*mode=*/1, angles, vbf, nullptr, NKV, T, HD);

  // attention -> (B*T, C) bf16
  attn_fwd<<<dim3(T / 64, NH, B), dim3(128), 0, stream>>>(qbf, kbf, vbf, abf, T);

  // output projection -> f32 d_out
  gemm_bf16_wmma<<<dim3(C / 128, M / 128), blk, 0, stream>>>(
      abf, woT, M, C, C, /*mode=*/2, angles, nullptr, outp, NH, T, HD);
}
